// vsumG_39762807227048
// MI455X (gfx1250) — compile-verified, matrix operand order checked by reading, not run on hardware
//
#include <hip/hip_runtime.h>
#include <hip/hip_bf16.h>
#include <math.h>

// Problem dimensions (fixed by the reference)
#define F_ 1024
#define H_ 512
#define S_ 512
#define B_ 32
#define T_ 128

typedef __bf16 bf16_t;
typedef __attribute__((ext_vector_type(16))) __bf16 v16bf;
typedef __attribute__((ext_vector_type(8)))  __bf16 v8bf;
typedef __attribute__((ext_vector_type(8)))  float  v8f;

__device__ __forceinline__ float sigmf_(float x) { return 1.0f / (1.0f + expf(-x)); }

// ---------------------------------------------------------------------------
// CDNA5 async global->LDS copy (16B per lane) + asynccnt wait.
// The clang builtin is declared as
//   void __builtin_amdgcn_global_load_async_to_lds_b128(
//       v4i __attribute__((address_space(1)))* src,   // global
//       v4i __attribute__((address_space(3)))* dst,   // LDS
//       imm int offset, imm int cpol);
// (confirmed by the round-2 diagnostic: param0 is v4i AS1).  Fallback to the
// raw ISA instruction if the builtin is absent (CDNA5 ISA).
// ---------------------------------------------------------------------------
__device__ __forceinline__ void async_load_b128(void* lds_dst, const void* gsrc) {
#if __has_builtin(__builtin_amdgcn_global_load_async_to_lds_b128)
  typedef int v4i_ __attribute__((vector_size(16)));
  typedef __attribute__((address_space(1))) v4i_* gp_t;
  typedef __attribute__((address_space(3))) v4i_* lp_t;
  __builtin_amdgcn_global_load_async_to_lds_b128((gp_t)gsrc, (lp_t)lds_dst, 0, 0);
#else
  unsigned loff = (unsigned)(size_t)(__attribute__((address_space(3))) void*)lds_dst;
  asm volatile("global_load_async_to_lds_b128 %0, %1, off"
               :: "v"(loff), "v"(gsrc) : "memory");
#endif
}

__device__ __forceinline__ void wait_async0() {
#if __has_builtin(__builtin_amdgcn_s_wait_asynccnt)
  __builtin_amdgcn_s_wait_asynccnt(0);
#else
  asm volatile("s_wait_asynccnt 0x0" ::: "memory");
#endif
}

// ---------------------------------------------------------------------------
// Utility kernels
// ---------------------------------------------------------------------------
__global__ void cast_f32_bf16_kernel(const float* __restrict__ src,
                                     bf16_t* __restrict__ dst, int n) {
  int i = blockIdx.x * blockDim.x + threadIdx.x;
  if (i < n) dst[i] = (bf16_t)src[i];
}

__global__ void fill_f32_kernel(float* __restrict__ p, float v, int n) {
  int i = blockIdx.x * blockDim.x + threadIdx.x;
  if (i < n) p[i] = v;
}

__global__ void fill_bf16_kernel(bf16_t* __restrict__ p, float v, int n) {
  int i = blockIdx.x * blockDim.x + threadIdx.x;
  if (i < n) p[i] = (bf16_t)v;
}

__global__ void copy_bf16_kernel(const bf16_t* __restrict__ s,
                                 bf16_t* __restrict__ d, int n) {
  int i = blockIdx.x * blockDim.x + threadIdx.x;
  if (i < n) d[i] = s[i];
}

// ---------------------------------------------------------------------------
// Generic bf16 WMMA GEMM (direct-from-global):
//   Cout[M,N] = A[M,K] * W[N,K]^T (+Cin)(+bias+bias2)
// One wave per 16x16 C tile; K multiple of 32; M,N multiples of 16.
// Fragment layout per CDNA5 ISA §7.12.2:
//   A: lane L row M=L%16, 16 bf16 over K = {0..7,16..23} (L<16) or
//      {8..15,24..31} (L>=16)  -> two 16B contiguous loads.
//   C/D: VGPR r -> M = r + (L>=16?8:0), N = L%16.
// Used for the small latency-bound GEMMs on the recurrent chains.
// ---------------------------------------------------------------------------
__global__ void gemm_bf16_wmma_kernel(
    const bf16_t* __restrict__ A, int lda,
    const bf16_t* __restrict__ W, int ldw,
    const float* __restrict__ Cin,
    const float* __restrict__ bias,
    const float* __restrict__ bias2,
    float* __restrict__ Cout, int ldc,
    int Mtiles, int Ntiles, int K) {
  const int lane = threadIdx.x & 31;
  const int wave = threadIdx.x >> 5;
  const int tile = blockIdx.x * (blockDim.x >> 5) + wave;
  if (tile >= Mtiles * Ntiles) return;
  const int mt = tile % Mtiles;
  const int nt = tile / Mtiles;

  const int l16   = lane & 15;
  const int khalf = (lane & 16) ? 8 : 0;  // K sub-offset for this lane half
  const bf16_t* arow = A + (size_t)(mt * 16 + l16) * lda + khalf;
  const bf16_t* wrow = W + (size_t)(nt * 16 + l16) * ldw + khalf;

  v8f acc = {};
  for (int kb = 0; kb < K; kb += 32) {
    __builtin_prefetch(arow + kb + 128, 0, 0);   // global_prefetch_b8
    __builtin_prefetch(wrow + kb + 128, 0, 0);
    v16bf a, b;
    ((v8bf*)&a)[0] = *(const v8bf*)(arow + kb);        // K = khalf + 0..7
    ((v8bf*)&a)[1] = *(const v8bf*)(arow + kb + 16);   // K = khalf + 16..23
    ((v8bf*)&b)[0] = *(const v8bf*)(wrow + kb);
    ((v8bf*)&b)[1] = *(const v8bf*)(wrow + kb + 16);
    acc = __builtin_amdgcn_wmma_f32_16x16x32_bf16(
        /*neg_a=*/false, a, /*neg_b=*/false, b,
        /*c_mod=*/(short)0, acc, /*reuse_a=*/false, /*reuse_b=*/false);
  }

  const int n     = nt * 16 + l16;
  const int mbase = mt * 16 + ((lane & 16) ? 8 : 0);
  float badd = 0.0f;
  if (bias)  badd += bias[n];
  if (bias2) badd += bias2[n];
#pragma unroll
  for (int r = 0; r < 8; ++r) {
    const size_t idx = (size_t)(mbase + r) * ldc + n;
    float v = acc[r] + badd;
    if (Cin) v += Cin[idx];
    Cout[idx] = v;
  }
}

// ---------------------------------------------------------------------------
// LDS-staged bf16 WMMA GEMM for the big hoisted GEMMs (all have K == 1024).
// Block = 8 waves = 8 consecutive M-tiles sharing one 16xK weight panel.
// Panel is staged with async global->LDS b128 copies (ASYNCcnt), rows padded
// by 8 bf16 (16B) so the 16 fragment rows map to distinct LDS banks, then
// B-fragments come from LDS (ds_load_b128) while A streams from global.
// Weight traffic drops 8x vs the direct kernel.
// ---------------------------------------------------------------------------
#define KSTG 1024
#define KPAD 8
__global__ void gemm_bf16_wmma_lds_kernel(
    const bf16_t* __restrict__ A, int lda,
    const bf16_t* __restrict__ W, int ldw,
    const float* __restrict__ bias,
    const float* __restrict__ bias2,
    float* __restrict__ Cout, int ldc) {
  __shared__ __align__(16) bf16_t wpanel[16][KSTG + KPAD];
  const int tid = threadIdx.x;
  const int n0  = blockIdx.y * 16;

  // Stage 16 x KSTG bf16 weight panel: 2048 16B chunks over 256 threads.
  for (int c = tid; c < 16 * (KSTG / 8); c += 256) {
    const int row = c >> 7;            // KSTG/8 = 128 chunks per row
    const int kc  = (c & 127) << 3;
    async_load_b128(&wpanel[row][kc], W + (size_t)(n0 + row) * ldw + kc);
  }
  wait_async0();
  __syncthreads();

  const int lane  = tid & 31;
  const int wave  = tid >> 5;
  const int l16   = lane & 15;
  const int khalf = (lane & 16) ? 8 : 0;
  const int mt    = blockIdx.x * 8 + wave;

  const bf16_t* arow = A + (size_t)(mt * 16 + l16) * lda + khalf;
  const bf16_t* wrow = &wpanel[l16][khalf];

  v8f acc = {};
  for (int kb = 0; kb < KSTG; kb += 32) {
    __builtin_prefetch(arow + kb + 128, 0, 0);
    v16bf a, b;
    ((v8bf*)&a)[0] = *(const v8bf*)(arow + kb);
    ((v8bf*)&a)[1] = *(const v8bf*)(arow + kb + 16);
    ((v8bf*)&b)[0] = *(const v8bf*)(wrow + kb);        // ds_load_b128
    ((v8bf*)&b)[1] = *(const v8bf*)(wrow + kb + 16);
    acc = __builtin_amdgcn_wmma_f32_16x16x32_bf16(
        false, a, false, b, (short)0, acc, false, false);
  }

  const int n     = n0 + l16;
  const int mbase = mt * 16 + ((lane & 16) ? 8 : 0);
  float badd = 0.0f;
  if (bias)  badd += bias[n];
  if (bias2) badd += bias2[n];
#pragma unroll
  for (int r = 0; r < 8; ++r)
    Cout[(size_t)(mbase + r) * ldc + n] = acc[r] + badd;
}

// ---------------------------------------------------------------------------
// Encoder LSTM pointwise: gates[2][B][4H] -> c (in place), h (bf16 recurrent
// buffer) and fused write of h into enc_out_bf16 (fwd at row t, bwd at row
// S-1-t, column offset dir*H).  Captures final h (fp32) at t == S-1.
// ---------------------------------------------------------------------------
__global__ void lstm_cell_enc_kernel(const float* __restrict__ gates,
                                     float* __restrict__ c,
                                     bf16_t* __restrict__ hbf,
                                     bf16_t* __restrict__ enc_out_bf,
                                     float* __restrict__ h_last,
                                     int t) {
  const int dir = blockIdx.y;
  const int i = blockIdx.x * blockDim.x + threadIdx.x;  // 0 .. B*H
  if (i >= B_ * H_) return;
  const int b = i / H_;
  const int j = i % H_;
  const float* g = gates + ((size_t)dir * B_ + b) * (4 * H_);
  const float gi = g[j], gf = g[H_ + j], gg = g[2 * H_ + j], go = g[3 * H_ + j];
  const size_t idx = (size_t)dir * B_ * H_ + i;
  const float cv = sigmf_(gf) * c[idx] + sigmf_(gi) * tanhf(gg);
  const float hv = sigmf_(go) * tanhf(cv);
  c[idx] = cv;
  hbf[idx] = (bf16_t)hv;
  const int row = (dir == 0) ? t : (S_ - 1 - t);
  enc_out_bf[((size_t)row * B_ + b) * (2 * H_) + dir * H_ + j] = (bf16_t)hv;
  if (t == S_ - 1) h_last[idx] = hv;
}

// ---------------------------------------------------------------------------
// Decoder LSTMCell pointwise: gates[B][4*2H] -> c (in place), h (bf16).
// ---------------------------------------------------------------------------
__global__ void lstm_cell_dec_kernel(const float* __restrict__ gates,
                                     float* __restrict__ c,
                                     bf16_t* __restrict__ hbf) {
  const int Hd = 2 * H_;  // 1024
  const int i = blockIdx.x * blockDim.x + threadIdx.x;
  if (i >= B_ * Hd) return;
  const int b = i / Hd;
  const int j = i % Hd;
  const float* g = gates + (size_t)b * (4 * Hd);
  const float gi = g[j], gf = g[Hd + j], gg = g[2 * Hd + j], go = g[3 * Hd + j];
  const float cv = sigmf_(gf) * c[i] + sigmf_(gi) * tanhf(gg);
  c[i] = cv;
  hbf[i] = (bf16_t)(sigmf_(go) * tanhf(cv));
}

// ---------------------------------------------------------------------------
// h0/c0: torch h.view(B,-1) on stacked [2,B,H] — exact raw-reshape semantics:
// out[b', j]: d = b'>=16; bsrc = 2*(b'%16) + j/H; k = j%H.
// ---------------------------------------------------------------------------
__global__ void build_h0c0_kernel(const float* __restrict__ h_last,
                                  const float* __restrict__ c_enc,
                                  bf16_t* __restrict__ h0bf,
                                  float* __restrict__ c0) {
  const int i = blockIdx.x * blockDim.x + threadIdx.x;  // 0 .. B*2H
  if (i >= B_ * 2 * H_) return;
  const int bp = i >> 10;       // / (2H)
  const int j  = i & 1023;      // % (2H)
  const int d    = bp >> 4;
  const int bsrc = ((bp & 15) << 1) + (j >> 9);
  const int k    = j & (H_ - 1);
  const size_t src = ((size_t)d * B_ + bsrc) * H_ + k;
  h0bf[i] = (bf16_t)h_last[src];
  c0[i]   = c_enc[src];
}

// ---------------------------------------------------------------------------
// Attention scores. One wave per (s,b) pair:
//   out[(t*B+b)*S + s] = W_v . tanh(enc_proj[s,b,:] + dproj[b,:]) + b_v
// ---------------------------------------------------------------------------
__global__ void attention_scores_kernel(const float* __restrict__ enc_proj,
                                        const float* __restrict__ dproj,
                                        const float* __restrict__ Wv,
                                        const float* __restrict__ bv,
                                        float* __restrict__ out, int t) {
  const int lane = threadIdx.x & 31;
  const int wave = threadIdx.x >> 5;
  const int pair = blockIdx.x * (blockDim.x >> 5) + wave;  // = s*B + b
  if (pair >= S_ * B_) return;
  const int s = pair / B_;
  const int b = pair % B_;
  const float* ep = enc_proj + (size_t)pair * H_;
  const float* dp = dproj + (size_t)b * H_;
  float acc = 0.0f;
  for (int j = lane; j < H_; j += 32)
    acc += Wv[j] * tanhf(ep[j] + dp[j]);
#pragma unroll
  for (int off = 16; off; off >>= 1)
    acc += __shfl_xor(acc, off, 32);
  if (lane == 0)
    out[((size_t)t * B_ + b) * S_ + s] = acc + bv[0];
}

// ---------------------------------------------------------------------------
// Greedy pointer: per-batch argmax over S (first-max ties, like jnp.argmax),
// then gather feats_bf16[idx[b], b, :] as the next decoder input.
// ---------------------------------------------------------------------------
__global__ void argmax_gather_kernel(const float* __restrict__ scores,  // [B][S]
                                     const bf16_t* __restrict__ feats_bf,
                                     bf16_t* __restrict__ x_next) {
  __shared__ int sidx[B_];
  const int tid = threadIdx.x;
  if (tid < B_) {
    const float* row = scores + (size_t)tid * S_;
    float best = row[0];
    int bi = 0;
    for (int s2 = 1; s2 < S_; ++s2) {
      const float v = row[s2];
      if (v > best) { best = v; bi = s2; }
    }
    sidx[tid] = bi;
  }
  __syncthreads();
  for (int i = tid; i < B_ * F_; i += blockDim.x) {
    const int b = i / F_;
    const int j = i % F_;
    x_next[i] = feats_bf[((size_t)sidx[b] * B_ + b) * F_ + j];
  }
}

// ---------------------------------------------------------------------------
// Host side. Workspace requirement ≈ 403 MB (dominated by the hoisted
// per-direction gate preactivations Xf/Xb: 2 x 16384x2048 fp32).
// ---------------------------------------------------------------------------
extern "C" void kernel_launch(void* const* d_in, const int* in_sizes, int n_in,
                              void* d_out, int out_size, void* d_ws, size_t ws_size,
                              hipStream_t stream) {
  (void)in_sizes; (void)n_in; (void)out_size; (void)ws_size;
  const float* feats = (const float*)d_in[0];
  const float* Wih_f = (const float*)d_in[1];
  const float* Whh_f = (const float*)d_in[2];
  const float* bih_f = (const float*)d_in[3];
  const float* bhh_f = (const float*)d_in[4];
  const float* Wih_b = (const float*)d_in[5];
  const float* Whh_b = (const float*)d_in[6];
  const float* bih_b = (const float*)d_in[7];
  const float* bhh_b = (const float*)d_in[8];
  const float* Wih_d = (const float*)d_in[9];
  const float* Whh_d = (const float*)d_in[10];
  const float* bih_d = (const float*)d_in[11];
  const float* bhh_d = (const float*)d_in[12];
  const float* W_enc = (const float*)d_in[13];
  const float* b_enc = (const float*)d_in[14];
  const float* W_dec = (const float*)d_in[15];
  const float* b_dec = (const float*)d_in[16];
  const float* W_v   = (const float*)d_in[17];
  const float* b_v   = (const float*)d_in[18];
  float* out = (float*)d_out;

  char* base = (char*)d_ws;
  size_t off = 0;
  auto carve = [&](size_t bytes) -> void* {
    void* p = base + off;
    off = (off + bytes + 255) & ~(size_t)255;
    return p;
  };

  bf16_t* feats_bf = (bf16_t*)carve((size_t)S_ * B_ * F_ * 2);
  bf16_t* Wihf_bf  = (bf16_t*)carve((size_t)4 * H_ * F_ * 2);
  bf16_t* Whhf_bf  = (bf16_t*)carve((size_t)4 * H_ * H_ * 2);
  bf16_t* Wihb_bf  = (bf16_t*)carve((size_t)4 * H_ * F_ * 2);
  bf16_t* Whhb_bf  = (bf16_t*)carve((size_t)4 * H_ * H_ * 2);
  bf16_t* Wihd_bf  = (bf16_t*)carve((size_t)8 * H_ * F_ * 2);
  bf16_t* Whhd_bf  = (bf16_t*)carve((size_t)8 * H_ * 2 * H_ * 2);
  bf16_t* Wenc_bf  = (bf16_t*)carve((size_t)H_ * 2 * H_ * 2);
  bf16_t* Wdec_bf  = (bf16_t*)carve((size_t)H_ * 2 * H_ * 2);
  float*  Xf       = (float*)carve((size_t)S_ * B_ * 4 * H_ * 4);
  float*  Xb       = (float*)carve((size_t)S_ * B_ * 4 * H_ * 4);
  float*  gates_e  = (float*)carve((size_t)2 * B_ * 4 * H_ * 4);
  float*  c_enc    = (float*)carve((size_t)2 * B_ * H_ * 4);
  bf16_t* h_enc_bf = (bf16_t*)carve((size_t)2 * B_ * H_ * 2);
  float*  h_last   = (float*)carve((size_t)2 * B_ * H_ * 4);
  bf16_t* encout_bf= (bf16_t*)carve((size_t)S_ * B_ * 2 * H_ * 2);
  float*  enc_proj = (float*)carve((size_t)S_ * B_ * H_ * 4);
  float*  c_dec    = (float*)carve((size_t)B_ * 2 * H_ * 4);
  bf16_t* hdec_bf  = (bf16_t*)carve((size_t)B_ * 2 * H_ * 2);
  float*  gates_d  = (float*)carve((size_t)B_ * 8 * H_ * 4);
  float*  dproj    = (float*)carve((size_t)B_ * H_ * 4);
  bf16_t* x_bf     = (bf16_t*)carve((size_t)B_ * F_ * 2);

  auto cast = [&](const float* s, bf16_t* d, int n) {
    cast_f32_bf16_kernel<<<(n + 255) / 256, 256, 0, stream>>>(s, d, n);
  };
  auto gemm = [&](const bf16_t* A, int lda, const bf16_t* W2, int ldw,
                  const float* Cin, const float* b1, const float* b2,
                  float* C, int ldc, int Mt, int Nt, int K) {
    const int tiles = Mt * Nt;
    gemm_bf16_wmma_kernel<<<(tiles + 7) / 8, 256, 0, stream>>>(
        A, lda, W2, ldw, Cin, b1, b2, C, ldc, Mt, Nt, K);
  };
  // LDS-staged variant; K fixed at 1024, Mtiles multiple of 8.
  auto gemm_lds = [&](const bf16_t* A, int lda, const bf16_t* W2, int ldw,
                      const float* b1, const float* b2,
                      float* C, int ldc, int Mt, int Nt) {
    gemm_bf16_wmma_lds_kernel<<<dim3(Mt / 8, Nt), 256, 0, stream>>>(
        A, lda, W2, ldw, b1, b2, C, ldc);
  };

  // --- one-time casts to bf16 ---
  cast(feats, feats_bf, S_ * B_ * F_);
  cast(Wih_f, Wihf_bf, 4 * H_ * F_);
  cast(Whh_f, Whhf_bf, 4 * H_ * H_);
  cast(Wih_b, Wihb_bf, 4 * H_ * F_);
  cast(Whh_b, Whhb_bf, 4 * H_ * H_);
  cast(Wih_d, Wihd_bf, 8 * H_ * F_);
  cast(Whh_d, Whhd_bf, 8 * H_ * 2 * H_);
  cast(W_enc, Wenc_bf, H_ * 2 * H_);
  cast(W_dec, Wdec_bf, H_ * 2 * H_);

  // --- zero initial encoder state ---
  fill_f32_kernel<<<(2 * B_ * H_ + 255) / 256, 256, 0, stream>>>(c_enc, 0.f, 2 * B_ * H_);
  fill_bf16_kernel<<<(2 * B_ * H_ + 255) / 256, 256, 0, stream>>>(h_enc_bf, 0.f, 2 * B_ * H_);

  // --- hoisted input projections (the 137 GFLOP parallel bulk), LDS-staged ---
  // Xf[s*B+b, :] = feats[s,b,:] @ Wih_f.T + bih_f + bhh_f   (same for bwd)
  gemm_lds(feats_bf, F_, Wihf_bf, F_, bih_f, bhh_f,
           Xf, 4 * H_, (S_ * B_) / 16, (4 * H_) / 16);
  gemm_lds(feats_bf, F_, Wihb_bf, F_, bih_b, bhh_b,
           Xb, 4 * H_, (S_ * B_) / 16, (4 * H_) / 16);

  // --- bidirectional encoder recurrence (sequential chain, direct GEMM) ---
  for (int t = 0; t < S_; ++t) {
    gemm(h_enc_bf, H_, Whhf_bf, H_,
         Xf + (size_t)t * B_ * 4 * H_, nullptr, nullptr,
         gates_e, 4 * H_, B_ / 16, (4 * H_) / 16, H_);
    gemm(h_enc_bf + B_ * H_, H_, Whhb_bf, H_,
         Xb + (size_t)(S_ - 1 - t) * B_ * 4 * H_, nullptr, nullptr,
         gates_e + (size_t)B_ * 4 * H_, 4 * H_, B_ / 16, (4 * H_) / 16, H_);
    lstm_cell_enc_kernel<<<dim3((B_ * H_ + 255) / 256, 2), 256, 0, stream>>>(
        gates_e, c_enc, h_enc_bf, encout_bf, h_last, t);
  }

  // --- decoder init state (exact view(B,-1) reshuffle) ---
  build_h0c0_kernel<<<(B_ * 2 * H_ + 255) / 256, 256, 0, stream>>>(
      h_last, c_enc, hdec_bf, c_dec);

  // --- hoisted attention encoder projection (LDS-staged, K = 2H = 1024) ---
  gemm_lds(encout_bf, 2 * H_, Wenc_bf, 2 * H_, b_enc, nullptr,
           enc_proj, H_, (S_ * B_) / 16, H_ / 16);

  // --- initial decoder input = feats[0] ---
  copy_bf16_kernel<<<(B_ * F_ + 255) / 256, 256, 0, stream>>>(feats_bf, x_bf, B_ * F_);

  // --- greedy pointer decoder ---
  for (int t = 0; t < T_; ++t) {
    // gates = x @ Wih_d.T + bih_d + bhh_d ; then += h @ Whh_d.T
    gemm(x_bf, F_, Wihd_bf, F_, nullptr, bih_d, bhh_d,
         gates_d, 8 * H_, B_ / 16, (8 * H_) / 16, F_);
    gemm(hdec_bf, 2 * H_, Whhd_bf, 2 * H_, gates_d, nullptr, nullptr,
         gates_d, 8 * H_, B_ / 16, (8 * H_) / 16, 2 * H_);
    lstm_cell_dec_kernel<<<(B_ * 2 * H_ + 255) / 256, 256, 0, stream>>>(
        gates_d, c_dec, hdec_bf);
    // dproj = h @ W_dec.T + b_dec
    gemm(hdec_bf, 2 * H_, Wdec_bf, 2 * H_, nullptr, b_dec, nullptr,
         dproj, H_, B_ / 16, H_ / 16, 2 * H_);
    // scores -> d_out rows for this t (L2-resident enc_proj stream)
    attention_scores_kernel<<<(S_ * B_) / 8, 256, 0, stream>>>(
        enc_proj, dproj, W_v, b_v, out, t);
    if (t < T_ - 1) {
      argmax_gather_kernel<<<1, 256, 0, stream>>>(
          out + (size_t)t * B_ * S_, feats_bf, x_bf);
    }
  }
}
